// StackedSTU1D_48550310314358
// MI455X (gfx1250) — compile-verified, matrix operand order checked by reading, not run on hardware
//
#include <hip/hip_runtime.h>
#include <hip/hip_bf16.h>
#include <math.h>

#define BB 16
#define INC 3
#define LL 2048
#define DD 256
#define KK 8
#define DEPTH_N 4
#define FF 1025
#define NN2 (2*FF)          // 2050 valid spectral columns (re/im interleaved)
#define NCT 17              // spectral column tiles (17*128 = 2176, padded)
#define EPSV 1e-5f
#define PI_F 3.14159265358979323846f

typedef __attribute__((ext_vector_type(16))) __bf16          v16bf;
typedef __attribute__((ext_vector_type(8)))  float           v8f;
typedef __attribute__((ext_vector_type(16))) unsigned short  us16;  // 32 B

__device__ __forceinline__ unsigned short f2bf(float f) {
  unsigned int u = __builtin_bit_cast(unsigned int, f);
  u += 0x7FFFu + ((u >> 16) & 1u);           // round-to-nearest-even
  return (unsigned short)(u >> 16);
}

// Fragment element map (ISA 7.12.2, 16-bit operands):
//   k = e + 8*((e>>3) + (lane>>4));  inverse: lane_hi = (k>>3)&1, e = (k&7)+8*(k>>4)

// ---------------------------------------------------------------------------
// Pointwise lift
// ---------------------------------------------------------------------------
__global__ __launch_bounds__(256)
void lift_k(const float* __restrict__ x, const float* __restrict__ lw,
            const float* __restrict__ lb, float* __restrict__ h) {
  long long idx = (long long)blockIdx.x * blockDim.x + threadIdx.x;
  if (idx >= (long long)BB * DD * LL) return;
  int t = (int)(idx % LL);
  long long r = idx / LL;
  int o = (int)(r % DD);
  int b = (int)(r / DD);
  const float* xb = x + (long long)b * INC * LL + t;
  float acc = lb[o];
  acc += lw[o*4+0] * xb[0] + lw[o*4+1] * xb[LL] + lw[o*4+2] * xb[2*LL];
  acc += lw[o*4+3] * ((float)t / (float)(LL - 1));
  h[idx] = acc;
}

// ---------------------------------------------------------------------------
// LayerNorm over channel dim d at each (b,t)
// ---------------------------------------------------------------------------
__global__ __launch_bounds__(256)
void ln_chan(const float* __restrict__ h, float* __restrict__ z,
             const float* __restrict__ gamma, const float* __restrict__ beta) {
  int idx = blockIdx.x * blockDim.x + threadIdx.x;
  if (idx >= BB * LL) return;
  int b = idx / LL, t = idx % LL;
  const float* hb = h + (long long)b * DD * LL + t;
  float s = 0.f, s2 = 0.f;
  for (int d = 0; d < DD; d++) {
    float v = hb[(long long)d * LL];
    s += v; s2 += v * v;
  }
  float mu = s * (1.0f / DD);
  float var = s2 * (1.0f / DD) - mu * mu;
  float rinv = rsqrtf(var + EPSV);
  float* zb = z + (long long)b * DD * LL + t;
  for (int d = 0; d < DD; d++) {
    float v = (hb[(long long)d * LL] - mu) * rinv;
    zb[(long long)d * LL] = v * gamma[d] + beta[d];
  }
}

// ---------------------------------------------------------------------------
// Stockham radix-2 FFT, length 2048, LDS twiddle table, one WG per row.
// ---------------------------------------------------------------------------
__global__ __launch_bounds__(256)
void fft2048_fwd(const float* __restrict__ z, float* __restrict__ Xf) {
  __shared__ float2 bufA[LL];
  __shared__ float2 bufB[LL];
  __shared__ float2 tw[LL / 2];           // e^{-i*pi*j/1024}
  long long row = blockIdx.x;
  const float* zr = z + row * LL;
  int t = threadIdx.x;
  for (int j = t; j < LL / 2; j += 256) {
    float sn, cs; __sincosf(-PI_F * (float)j * (1.0f / (LL / 2)), &sn, &cs);
    tw[j].x = cs; tw[j].y = sn;
  }
  for (int i = t; i < LL; i += 256) { bufA[i].x = zr[i]; bufA[i].y = 0.f; }
  __syncthreads();
  float2* src = bufA; float2* dst = bufB;
  int sh = 10;
  for (int Ns = 1; Ns < LL; Ns <<= 1, sh--) {
    for (int j = t; j < LL / 2; j += 256) {
      float2 v0 = src[j];
      float2 v1 = src[j + LL / 2];
      int r = j & (Ns - 1);
      float2 w = tw[r << sh];
      float tr = w.x * v1.x - w.y * v1.y;
      float ti = w.x * v1.y + w.y * v1.x;
      int idxD = ((j ^ r) << 1) | r;
      dst[idxD].x = v0.x + tr;  dst[idxD].y = v0.y + ti;
      dst[idxD + Ns].x = v0.x - tr;  dst[idxD + Ns].y = v0.y - ti;
    }
    __syncthreads();
    float2* tmp = src; src = dst; dst = tmp;
  }
  float2* out = (float2*)(Xf + row * (long long)FF * 2);
  for (int f = t; f < FF; f += 256) out[f] = src[f];
}

// Inverse FFT of hermitian half-spectrum -> real signal (bf16 rows).
__global__ __launch_bounds__(256)
void ifft2048_real(const float* __restrict__ Sf, unsigned short* __restrict__ Sb) {
  __shared__ float2 bufA[LL];
  __shared__ float2 bufB[LL];
  __shared__ float2 tw[LL / 2];
  long long row = blockIdx.x;
  const float* in = Sf + row * NN2;
  int t = threadIdx.x;
  for (int j = t; j < LL / 2; j += 256) {
    float sn, cs; __sincosf(PI_F * (float)j * (1.0f / (LL / 2)), &sn, &cs);
    tw[j].x = cs; tw[j].y = sn;
  }
  for (int i = t; i < LL; i += 256) {
    float2 v;
    if (i < FF) { v = *(const float2*)(in + 2 * i); }
    else { int m = LL - i; v = *(const float2*)(in + 2 * m); v.y = -v.y; }
    bufA[i] = v;
  }
  __syncthreads();
  float2* src = bufA; float2* dst = bufB;
  int sh = 10;
  for (int Ns = 1; Ns < LL; Ns <<= 1, sh--) {
    for (int j = t; j < LL / 2; j += 256) {
      float2 v0 = src[j];
      float2 v1 = src[j + LL / 2];
      int r = j & (Ns - 1);
      float2 w = tw[r << sh];
      float tr = w.x * v1.x - w.y * v1.y;
      float ti = w.x * v1.y + w.y * v1.x;
      int idxD = ((j ^ r) << 1) | r;
      dst[idxD].x = v0.x + tr;  dst[idxD].y = v0.y + ti;
      dst[idxD + Ns].x = v0.x - tr;  dst[idxD + Ns].y = v0.y - ti;
    }
    __syncthreads();
    float2* tmp = src; src = dst; dst = tmp;
  }
  unsigned int* out = (unsigned int*)(Sb + row * LL);
  const float invN = 1.0f / (float)LL;
  for (int i = 2 * t; i < LL; i += 512) {
    unsigned int lo = f2bf(src[i].x * invN);
    unsigned int hi = f2bf(src[i + 1].x * invN);
    out[i >> 1] = lo | (hi << 16);
  }
}

// ---------------------------------------------------------------------------
// build_u2 (gather form): writes U2 directly in fragment-major layout
//   U2sw[b][kt(64)][ct(17)][frag(8)][lane(32)][e(16)]
// One thread produces one contiguous 32B us16 run. Pad cols (>=2050) -> 0.
// ---------------------------------------------------------------------------
__global__ __launch_bounds__(256)
void build_u2(const float* __restrict__ Xf, const float* __restrict__ pr,
              const float* __restrict__ pi_, unsigned short* __restrict__ U2sw) {
  long long u = (long long)blockIdx.x * blockDim.x + threadIdx.x;
  if (u >= (long long)BB * 64 * NCT * 8 * 32) return;
  int lane = (int)(u & 31);
  int fr   = (int)((u >> 5) & 7);
  long long rest = u >> 8;
  int ct = (int)(rest % NCT); rest /= NCT;
  int kt = (int)(rest & 63);
  int b  = (int)(rest >> 6);

  int n = ct * 128 + fr * 16 + (lane & 15);
  us16 val;
  if (n >= NN2) {
    #pragma unroll
    for (int e = 0; e < 16; e++) val[e] = 0;
  } else {
    int f = n >> 1, comp = n & 1;
    int l = kt >> 3;
    float p = pr[l * FF + f], q = pi_[l * FF + f];
    const float* xb = Xf + (long long)b * DD * FF * 2 + 2 * f;
    int lhi = lane >> 4;
    #pragma unroll
    for (int e = 0; e < 16; e++) {
      int k = kt * 32 + e + 8 * ((e >> 3) + lhi);
      int d = k & 255;
      float2 xv = *(const float2*)(xb + (long long)d * (FF * 2));
      float res = (comp == 0) ? (xv.x * p + xv.y * q)    // Re(Xf * conj(phi))
                              : (xv.y * p - xv.x * q);   // Im
      val[e] = f2bf(res);
    }
  }
  ((us16*)U2sw)[u] = val;
}

// ---------------------------------------------------------------------------
// Repack ifft rows into fragment-major layout:
//   Sbs[b][kt(8)][ct(16)][frag(8)][lane(32)][e(16)]
// ---------------------------------------------------------------------------
__global__ __launch_bounds__(256)
void sb_swizzle(const unsigned short* __restrict__ Sb,
                unsigned short* __restrict__ Sbs) {
  int u = blockIdx.x * blockDim.x + threadIdx.x;
  if (u >= BB * 8 * 16 * 8 * 32) return;
  int lane = u & 31;
  int fr   = (u >> 5) & 7;
  int ct   = (u >> 8) & 15;
  int kt   = (u >> 12) & 7;
  int b    = u >> 15;
  int n = ct * 128 + fr * 16 + (lane & 15);
  int lhi = lane >> 4;
  const unsigned short* sp = Sb + (long long)b * DD * LL + n;
  us16 val;
  #pragma unroll
  for (int e = 0; e < 16; e++) {
    int k = kt * 32 + e + 8 * ((e >> 3) + lhi);
    val[e] = sp[(long long)k * LL];
  }
  ((us16*)Sbs)[u] = val;
}

// ---------------------------------------------------------------------------
// Spectral GEMM, LDS-free: Sf(256 x 2050) = theta @ U2.
// Operands pre-swizzled; fragments loaded straight from global (L2).
// 8 waves (2x4), wave tile 64x32, 64 k-tiles, 8 WMMAs per tile. No barriers.
// ---------------------------------------------------------------------------
__global__ __launch_bounds__(256)
void gemm_spectral_direct(const unsigned short* __restrict__ Asw,
                          const unsigned short* __restrict__ Bsw,
                          float* __restrict__ C) {
  const int t = threadIdx.x;
  const int lane = t & 31, wid = t >> 5;
  const int wm = (wid >> 2) * 64;
  const int wn = (wid & 3) * 32;
  const int rowbase = blockIdx.y * 128;
  const int ct = blockIdx.x;
  float* Cp = C + (long long)blockIdx.z * DD * NN2;

  const us16* Ap = (const us16*)Asw + ((rowbase >> 4) + (wm >> 4)) * 32 + lane;
  const us16* Bp = (const us16*)Bsw +
      ((long long)blockIdx.z * 64 * NCT + ct) * 256 + (wn >> 4) * 32 + lane;
  const int A_STEP = 16 * 32;        // us16 per k-tile
  const int B_STEP = NCT * 256;      // us16 per k-tile

  v8f zero = {};
  v8f acc[4][2];
  #pragma unroll
  for (int i = 0; i < 4; i++)
    #pragma unroll
    for (int j = 0; j < 2; j++) acc[i][j] = zero;

  v16bf ca[4], cb[2];
  #pragma unroll
  for (int mi = 0; mi < 4; mi++) ca[mi] = __builtin_bit_cast(v16bf, Ap[mi * 32]);
  #pragma unroll
  for (int ni = 0; ni < 2; ni++) cb[ni] = __builtin_bit_cast(v16bf, Bp[ni * 32]);

  #pragma unroll 4
  for (int kt = 0; kt < 64; kt++) {
    v16bf na[4], nb[2];
    if (kt < 63) {
      const us16* An = Ap + (kt + 1) * A_STEP;
      const us16* Bn = Bp + (long long)(kt + 1) * B_STEP;
      if (kt < 62) __builtin_prefetch(Bp + (long long)(kt + 2) * B_STEP, 0, 1);
      #pragma unroll
      for (int mi = 0; mi < 4; mi++) na[mi] = __builtin_bit_cast(v16bf, An[mi * 32]);
      #pragma unroll
      for (int ni = 0; ni < 2; ni++) nb[ni] = __builtin_bit_cast(v16bf, Bn[ni * 32]);
    }
    #pragma unroll
    for (int mi = 0; mi < 4; mi++)
      #pragma unroll
      for (int ni = 0; ni < 2; ni++)
        acc[mi][ni] = __builtin_amdgcn_wmma_f32_16x16x32_bf16(
            false, ca[mi], false, cb[ni], (short)0, acc[mi][ni], false, false);
    if (kt < 63) {
      #pragma unroll
      for (int mi = 0; mi < 4; mi++) ca[mi] = na[mi];
      #pragma unroll
      for (int ni = 0; ni < 2; ni++) cb[ni] = nb[ni];
    }
  }

  const int colbase = ct * 128;
  #pragma unroll
  for (int mi = 0; mi < 4; mi++) {
    #pragma unroll
    for (int ni = 0; ni < 2; ni++) {
      int n = colbase + wn + ni * 16 + (lane & 15);
      if (n < NN2) {
        #pragma unroll
        for (int v = 0; v < 8; v++) {
          int m = rowbase + wm + mi * 16 + v + 8 * (lane >> 4);
          Cp[(long long)m * NN2 + n] = acc[mi][ni][v];
        }
      }
    }
  }
}

// ---------------------------------------------------------------------------
// Fused GLU, LDS-free: V = Wv@S, G = Wg@S (16 WMMAs per k-tile), then
// h += (V + b_v) * sigmoid(G + b_g). K = 256 -> 8 k-tiles, fully unrolled.
// ---------------------------------------------------------------------------
__global__ __launch_bounds__(256)
void glu_gemm_fused(const unsigned short* __restrict__ Wsw,   // [val;gate] swizzled
                    const unsigned short* __restrict__ Sbs,
                    float* __restrict__ H,
                    const float* __restrict__ vb, const float* __restrict__ gb) {
  const int t = threadIdx.x;
  const int lane = t & 31, wid = t >> 5;
  const int wm = (wid >> 2) * 64;
  const int wn = (wid & 3) * 32;
  const int rowbase = blockIdx.y * 128;
  const int ct = blockIdx.x;
  float* Hp = H + (long long)blockIdx.z * DD * LL;

  const us16* Avp = (const us16*)Wsw + ((rowbase >> 4) + (wm >> 4)) * 32 + lane;
  const us16* Agp = Avp + 8 * 16 * 32;                 // gate matrix offset
  const us16* Bp  = (const us16*)Sbs +
      ((long long)blockIdx.z * 8 * 16 + ct) * 256 + (wn >> 4) * 32 + lane;
  const int A_STEP = 16 * 32;
  const int B_STEP = 16 * 256;

  v8f zero = {};
  v8f accV[4][2], accG[4][2];
  #pragma unroll
  for (int i = 0; i < 4; i++)
    #pragma unroll
    for (int j = 0; j < 2; j++) { accV[i][j] = zero; accG[i][j] = zero; }

  #pragma unroll
  for (int kt = 0; kt < 8; kt++) {
    v16bf bfrag[2];
    #pragma unroll
    for (int ni = 0; ni < 2; ni++)
      bfrag[ni] = __builtin_bit_cast(v16bf, Bp[(long long)kt * B_STEP + ni * 32]);
    v16bf av[4], ag[4];
    #pragma unroll
    for (int mi = 0; mi < 4; mi++) {
      av[mi] = __builtin_bit_cast(v16bf, Avp[kt * A_STEP + mi * 32]);
      ag[mi] = __builtin_bit_cast(v16bf, Agp[kt * A_STEP + mi * 32]);
    }
    #pragma unroll
    for (int mi = 0; mi < 4; mi++)
      #pragma unroll
      for (int ni = 0; ni < 2; ni++)
        accV[mi][ni] = __builtin_amdgcn_wmma_f32_16x16x32_bf16(
            false, av[mi], false, bfrag[ni], (short)0, accV[mi][ni], false, false);
    #pragma unroll
    for (int mi = 0; mi < 4; mi++)
      #pragma unroll
      for (int ni = 0; ni < 2; ni++)
        accG[mi][ni] = __builtin_amdgcn_wmma_f32_16x16x32_bf16(
            false, ag[mi], false, bfrag[ni], (short)0, accG[mi][ni], false, false);
  }

  const int colbase = ct * 128;
  #pragma unroll
  for (int mi = 0; mi < 4; mi++) {
    #pragma unroll
    for (int ni = 0; ni < 2; ni++) {
      int n = colbase + wn + ni * 16 + (lane & 15);
      #pragma unroll
      for (int v = 0; v < 8; v++) {
        int m = rowbase + wm + mi * 16 + v + 8 * (lane >> 4);
        float vv = accV[mi][ni][v] + vb[m];
        float gg = accG[mi][ni][v] + gb[m];
        float* hp = Hp + (long long)m * LL + n;
        *hp += vv * (1.0f / (1.0f + __expf(-gg)));
      }
    }
  }
}

// ---------------------------------------------------------------------------
// Head
// ---------------------------------------------------------------------------
__global__ __launch_bounds__(256)
void head_k(const float* __restrict__ h, const float* __restrict__ hw,
            const float* __restrict__ hb, float* __restrict__ y) {
  int idx = blockIdx.x * blockDim.x + threadIdx.x;
  if (idx >= BB * LL) return;
  int b = idx / LL, t = idx % LL;
  const float* hp = h + (long long)b * DD * LL + t;
  float acc = hb[0];
  for (int d = 0; d < DD; d++) acc += hw[d] * hp[(long long)d * LL];
  y[idx] = acc;
}

// ---------------------------------------------------------------------------
// Weight repack to bf16, fragment-major.
// Asw: [kt(64)][rf(16)][lane(32)][e(16)]  (theta as 256 x 2048)
// ---------------------------------------------------------------------------
__global__ __launch_bounds__(256)
void conv_theta(const float* __restrict__ th, unsigned short* __restrict__ Asw) {
  int idx = blockIdx.x * blockDim.x + threadIdx.x;
  if (idx >= DD * KK * DD) return;
  int kidx = idx % (KK * DD);           // (l,d)
  int hrow = idx / (KK * DD);
  int l = kidx / DD, d = kidx % DD;
  int k = kidx;                         // l*256 + d
  int kt = k >> 5, kk = k & 31;
  int lane = (hrow & 15) + 16 * ((kk >> 3) & 1);
  int e = (kk & 7) + 8 * (kk >> 4);
  int rf = hrow >> 4;
  Asw[((kt * 16 + rf) * 32 + lane) * 16 + e] =
      f2bf(th[((long long)l * DD + hrow) * DD + d]);
}

// Wsw: [which(2)][kt(8)][rf(16)][lane(32)][e(16)]  (val then gate, 256x256)
__global__ __launch_bounds__(256)
void conv_glu(const float* __restrict__ vw, const float* __restrict__ gw,
              unsigned short* __restrict__ Wsw) {
  int idx = blockIdx.x * blockDim.x + threadIdx.x;
  if (idx >= 2 * DD * DD) return;
  int d = idx % DD;                     // k dim
  int r = idx / DD;                     // row (0..511)
  int which = r >> 8;
  int m = r & 255;
  int kt = d >> 5, kk = d & 31;
  int lane = (m & 15) + 16 * ((kk >> 3) & 1);
  int e = (kk & 7) + 8 * (kk >> 4);
  int rf = m >> 4;
  float val = which ? gw[(long long)m * DD + d] : vw[(long long)m * DD + d];
  Wsw[which * (8 * 16 * 32 * 16) + ((kt * 16 + rf) * 32 + lane) * 16 + e] = f2bf(val);
}

// ---------------------------------------------------------------------------
extern "C" void kernel_launch(void* const* d_in, const int* in_sizes, int n_in,
                              void* d_out, int out_size, void* d_ws, size_t ws_size,
                              hipStream_t stream) {
  (void)in_sizes; (void)n_in; (void)out_size; (void)ws_size;
  const float* x      = (const float*)d_in[0];
  const float* lift_w = (const float*)d_in[1];
  const float* lift_b = (const float*)d_in[2];
  const float* theta  = (const float*)d_in[3];
  const float* ln_g   = (const float*)d_in[4];
  const float* ln_b   = (const float*)d_in[5];
  const float* gvw    = (const float*)d_in[6];
  const float* gvb    = (const float*)d_in[7];
  const float* ggw    = (const float*)d_in[8];
  const float* ggb    = (const float*)d_in[9];
  const float* hw     = (const float*)d_in[10];
  const float* hb     = (const float*)d_in[11];
  const float* phir   = (const float*)d_in[12];
  const float* phii   = (const float*)d_in[13];
  float* out = (float*)d_out;

  char* ws = (char*)d_ws;
  size_t cur = 0;
  auto alloc = [&](size_t bytes) -> char* {
    char* p = ws + cur;
    cur += (bytes + 255) & ~(size_t)255;
    return p;
  };
  float* h  = (float*)alloc((size_t)BB * DD * LL * 4);             // 33.5 MB
  float* z  = (float*)alloc((size_t)BB * DD * LL * 4);             // 33.5 MB
  float* Xf = (float*)alloc((size_t)BB * DD * FF * 2 * 4);         // 33.6 MB
  float* Sf = (float*)alloc((size_t)BB * DD * NN2 * 4);            // 33.6 MB
  unsigned short* U2s = (unsigned short*)alloc((size_t)BB * 64 * NCT * 4096 * 2); // 143 MB
  unsigned short* Sb  = (unsigned short*)alloc((size_t)BB * DD * LL * 2);         // 16.8 MB
  unsigned short* Sbs = (unsigned short*)alloc((size_t)BB * 8 * 16 * 4096 * 2);   // 16.8 MB
  unsigned short* Asw = (unsigned short*)alloc((size_t)64 * 16 * 32 * 16 * 2);    // 1 MB
  unsigned short* Wsw = (unsigned short*)alloc((size_t)2 * 8 * 16 * 32 * 16 * 2); // 256 KB

  dim3 blk(256);
  lift_k<<<dim3((unsigned)(((long long)BB * DD * LL + 255) / 256)), blk, 0, stream>>>(
      x, lift_w, lift_b, h);

  const long long nU2 = (long long)BB * 64 * NCT * 8 * 32;   // us16 units
  for (int i = 0; i < DEPTH_N; i++) {
    conv_theta<<<(DD * KK * DD + 255) / 256, blk, 0, stream>>>(
        theta + (size_t)i * KK * DD * DD, Asw);
    conv_glu<<<(2 * DD * DD + 255) / 256, blk, 0, stream>>>(
        gvw + (size_t)i * DD * DD, ggw + (size_t)i * DD * DD, Wsw);
    ln_chan<<<(BB * LL + 255) / 256, blk, 0, stream>>>(
        h, z, ln_g + i * DD, ln_b + i * DD);
    fft2048_fwd<<<BB * DD, blk, 0, stream>>>(z, Xf);
    build_u2<<<(unsigned)((nU2 + 255) / 256), blk, 0, stream>>>(
        Xf, phir, phii, U2s);
    gemm_spectral_direct<<<dim3(NCT, DD / 128, BB), blk, 0, stream>>>(
        Asw, U2s, Sf);
    ifft2048_real<<<BB * DD, blk, 0, stream>>>(Sf, Sb);
    sb_swizzle<<<(BB * 8 * 16 * 8 * 32 + 255) / 256, blk, 0, stream>>>(Sb, Sbs);
    glu_gemm_fused<<<dim3(LL / 128, DD / 128, BB), blk, 0, stream>>>(
        Wsw, Sbs, h, gvb + i * DD, ggb + i * DD);
  }
  head_k<<<(BB * LL + 255) / 256, blk, 0, stream>>>(h, hw, hb, out);
}
